// AlternationN_Torch_34093450395771
// MI455X (gfx1250) — matrix operand order repaired and checked by reading, not compiled, not run on hardware
//
#include <hip/hip_runtime.h>
#include <hip/hip_bf16.h>
#include <math.h>

// ---- problem dims ----
#define T_ 128
#define B_ 256
#define H_ 128
#define KD_ 64
#define VD_ 128
#define O_ 512
#define DM_ 32

typedef __attribute__((ext_vector_type(16))) __bf16 v16bf;
typedef __attribute__((ext_vector_type(8)))  float  v8f;

// ---------------- WMMA helpers (CDNA5 16x16x32 bf16, f32 accum) ----------------
__device__ __forceinline__ v8f wmma_bf16(v16bf a, v16bf b, v8f c) {
  return __builtin_amdgcn_wmma_f32_16x16x32_bf16(false, a, false, b, (short)0, c,
                                                 false, false);
}

// A-matrix 16x32 bf16 fragment from row-major A[lda], tile origin (m0,k0).
// ISA 7.12.2: lanes 0-15 M=0..15 K in {0..7,16..23}; lanes 16-31 same M, K in {8..15,24..31}
__device__ __forceinline__ v16bf load_a_frag(const __bf16* __restrict__ A, int lda,
                                             int m0, int k0, int lane) {
  const int half = lane >> 4;
  const int mr   = lane & 15;
  const __bf16* row = A + (size_t)(m0 + mr) * lda + k0;
  v16bf a;
#pragma unroll
  for (int e = 0; e < 16; ++e) {
    int v = e >> 1, p = e & 1;
    int k = (v < 4) ? ((half ? 8 : 0) + 2 * v + p)
                    : ((half ? 24 : 16) + 2 * (v - 4) + p);
    a[e] = row[k];
  }
  return a;
}

// B-fragment from pre-swizzled (fragment-major) weight buffer: one contiguous
// 32-byte load per lane, fully coalesced per wave.
__device__ __forceinline__ v16bf load_b_swz(const __bf16* __restrict__ Bswz,
                                            int KT, int kt, int nt, int lane) {
  return *(const v16bf*)(Bswz + ((((size_t)nt * KT + kt) * 32 + lane) << 4));
}

// map fragment-layout flat index -> (k, n) of the logical KdimxNdim B matrix
__device__ __forceinline__ void swz_to_kn(int r, int KT, int* k, int* n) {
  int e = r & 15;
  int lane = (r >> 4) & 31;
  int frag = r >> 9;
  int kt = frag % KT, nt = frag / KT;
  *k = kt * 32 + ((lane >> 4) ? 8 : 0) + (e & 7) + ((e >> 3) ? 16 : 0);
  *n = nt * 16 + (lane & 15);
}

// ---------------- prep: repack weights to bf16 fragment-major layouts ----------------
// W2p:    logical [k=j(128), n=i*32+d(4096)] = hyp_W2[(i*128+j)*32+d],  KT=4
// predWT: logical [k=v(128), n=o(512)]       = pred_W[o*128+v],         KT=4
// gateWT: logical [k=j2(256), n=h(128)]      = gate_W[h*256+j2],        KT=8
// kvWT:   logical [k=j(128), n=c(192)]       = c<64?Wqk[c,j]:Wv[c-64,j],KT=4
__global__ void k_prep_weights(const float* __restrict__ hyp_W2,
                               const float* __restrict__ pred_W,
                               const float* __restrict__ gate_W,
                               const float* __restrict__ Wqk,
                               const float* __restrict__ Wv,
                               __bf16* __restrict__ W2p, __bf16* __restrict__ predWT,
                               __bf16* __restrict__ gateWT, __bf16* __restrict__ kvWT) {
  int idx = blockIdx.x * blockDim.x + threadIdx.x;
  const int N0 = 128 * 4096, N1 = 128 * 512, N2 = 256 * 128, N3 = 128 * 192;
  int k, n;
  if (idx < N0) {
    swz_to_kn(idx, 4, &k, &n);
    W2p[idx] = (__bf16)hyp_W2[(size_t)((n >> 5) * 128 + k) * 32 + (n & 31)];
  } else if (idx < N0 + N1) {
    int r = idx - N0;
    swz_to_kn(r, 4, &k, &n);
    predWT[r] = (__bf16)pred_W[(size_t)n * 128 + k];
  } else if (idx < N0 + N1 + N2) {
    int r = idx - N0 - N1;
    swz_to_kn(r, 8, &k, &n);
    gateWT[r] = (__bf16)gate_W[(size_t)n * 256 + k];
  } else if (idx < N0 + N1 + N2 + N3) {
    int r = idx - N0 - N1 - N2;
    swz_to_kn(r, 4, &k, &n);
    kvWT[r] = (__bf16)((n < 64) ? Wqk[(size_t)n * 128 + k] : Wv[(size_t)(n - 64) * 128 + k]);
  }
}

// vemb_all[t,b,h], hyp_all[t,b,d]=silu(...), hidden init
__global__ void k_prep_state(const float* __restrict__ vel, const float* __restrict__ vel_W,
                             const float* __restrict__ vel_b, const float* __restrict__ hyp_W1,
                             const float* __restrict__ hyp_b1, const float* __restrict__ hid0,
                             float* __restrict__ vemb_all, float* __restrict__ hyp_all,
                             float* __restrict__ hidden, __bf16* __restrict__ hidden_bf) {
  int tb = blockIdx.x, h = threadIdx.x;
  float v0 = vel[tb * 3 + 0], v1 = vel[tb * 3 + 1], v2 = vel[tb * 3 + 2];
  vemb_all[(size_t)tb * H_ + h] =
      vel_W[h * 3 + 0] * v0 + vel_W[h * 3 + 1] * v1 + vel_W[h * 3 + 2] * v2 + vel_b[h];
  if (h < DM_) {
    float x = hyp_W1[h * 3 + 0] * v0 + hyp_W1[h * 3 + 1] * v1 + hyp_W1[h * 3 + 2] * v2 + hyp_b1[h];
    hyp_all[(size_t)tb * DM_ + h] = x / (1.0f + expf(-x));
  }
  if (tb < B_) {
    float hv = tanhf(hid0[h]);
    hidden[tb * H_ + h]    = hv;
    hidden_bf[tb * H_ + h] = (__bf16)hv;
  }
}

// ---------------- step kernel 1: U[256,4096] = hidden[256,128] @ W2p[128,4096] ----------------
// wave = 16x16 m-tile x 4 consecutive n-tiles; A frags loaded once, 16 WMMAs/wave.
__global__ void k_hyper_gemm(const __bf16* __restrict__ hid_bf, const __bf16* __restrict__ W2p,
                             float* __restrict__ U) {
  const int lane = threadIdx.x & 31, wave = threadIdx.x >> 5;
  const int w = blockIdx.x * 8 + wave;        // 1024 waves
  const int m0 = (w >> 6) * 16, nt0 = (w & 63) * 4;
  v16bf a[4];
#pragma unroll
  for (int kt = 0; kt < 4; ++kt) a[kt] = load_a_frag(hid_bf, H_, m0, kt * 32, lane);
  const int nn_l = lane & 15;
  const int mb   = m0 + ((lane >> 4) ? 8 : 0);
#pragma unroll
  for (int s = 0; s < 4; ++s) {
    const int nt = nt0 + s;
    v8f acc = {};
#pragma unroll
    for (int kt = 0; kt < 4; ++kt)
      acc = wmma_bf16(a[kt], load_b_swz(W2p, 4, kt, nt, lane), acc);
    const int nn = nt * 16 + nn_l;
#pragma unroll
    for (int v = 0; v < 8; ++v) U[(size_t)(mb + v) * 4096 + nn] = acc[v];
  }
}

// ---------------- step kernel 2: pi, tanh+LN -> pi_pred, pack gate A ----------------
__global__ void k_pi(const float* __restrict__ U, const float* __restrict__ hyp_all,
                     const float* __restrict__ hidden, const float* __restrict__ hyp_b2,
                     const float* __restrict__ vemb_all, const int* __restrict__ obs,
                     const float* __restrict__ embed_W, float* __restrict__ pi_ws,
                     float* __restrict__ pi_pred_ws, __bf16* __restrict__ Agate,
                     float* __restrict__ gen_out, int t) {
  __shared__ float sh_hid[H_];
  __shared__ float sh_hyp[DM_];
  __shared__ float red[H_];
  const int b = blockIdx.x, ir = threadIdx.x;
  sh_hid[ir] = hidden[b * H_ + ir];
  if (ir < DM_) sh_hyp[ir] = hyp_all[(size_t)(t * B_ + b) * DM_ + ir];
  __syncthreads();
  float s = 0.f;
  const float* Ub = U + (size_t)b * 4096 + ir * DM_;
#pragma unroll
  for (int d = 0; d < DM_; ++d) s += Ub[d] * sh_hyp[d];
  const float* b2 = hyp_b2 + (size_t)ir * H_;
  for (int j = 0; j < H_; ++j) s += b2[j] * sh_hid[j];
  s += vemb_all[(size_t)(t * B_ + b) * H_ + ir];
  pi_ws[b * H_ + ir] = s;
  gen_out[(size_t)(t * B_ + b) * H_ + ir] = s;
  Agate[b * 256 + ir]       = (__bf16)s;
  Agate[b * 256 + 128 + ir] = (__bf16)embed_W[(size_t)obs[t * B_ + b] * H_ + ir];
  float tn = tanhf(s);
  red[ir] = tn; __syncthreads();
  for (int off = 64; off > 0; off >>= 1) { if (ir < off) red[ir] += red[ir + off]; __syncthreads(); }
  float mean = red[0] * (1.0f / H_); __syncthreads();
  float dv = tn - mean;
  red[ir] = dv * dv; __syncthreads();
  for (int off = 64; off > 0; off >>= 1) { if (ir < off) red[ir] += red[ir + off]; __syncthreads(); }
  float var = red[0] * (1.0f / H_);
  pi_pred_ws[b * H_ + ir] = dv * rsqrtf(var + 1e-5f);
}

// ---------------- step kernel 3: attention + memory readout + write MLP ----------------
__global__ void k_attn(const float* __restrict__ pi_pred, const float* __restrict__ Wqk,
                       const float* __restrict__ keys, const float* __restrict__ vals,
                       float* __restrict__ probs, const float* __restrict__ expl,
                       const float* __restrict__ mem_W1, const float* __restrict__ mem_b1,
                       const float* __restrict__ mem_W2, const float* __restrict__ mem_b2,
                       float* __restrict__ mem_ws, __bf16* __restrict__ mem_bf, int t) {
  __shared__ float sh_pi[H_];
  __shared__ float sh_q[KD_];
  __shared__ float sh_c[T_];
  __shared__ float red[T_];
  __shared__ float sh_h1[200];
  const int b = blockIdx.x, tid = threadIdx.x;
  sh_pi[tid] = pi_pred[b * H_ + tid];
  __syncthreads();
  if (tid < KD_) {
    float s = 0.f;
    const float* w = Wqk + (size_t)tid * H_;
#pragma unroll 8
    for (int j = 0; j < H_; ++j) s += w[j] * sh_pi[j];
    sh_q[tid] = s * 0.125f;  // 1/sqrt(K)
  }
  __syncthreads();
  float score = -1e9f;
  if (tid < t) {
    const float* kr = keys + (size_t)(tid * B_ + b) * KD_;
    float s = 0.f;
#pragma unroll 8
    for (int k = 0; k < KD_; ++k) s += sh_q[k] * kr[k];
    score = s;
  }
  red[tid] = score; __syncthreads();
  for (int off = 64; off > 0; off >>= 1) { if (tid < off) red[tid] = fmaxf(red[tid], red[tid + off]); __syncthreads(); }
  float smax = red[0]; __syncthreads();
  float w = expf(score - smax);
  red[tid] = w; __syncthreads();
  for (int off = 64; off > 0; off >>= 1) { if (tid < off) red[tid] += red[tid + off]; __syncthreads(); }
  float denom = red[0]; __syncthreads();
  sh_c[tid] = (tid < t) ? (w / denom) * probs[tid * B_ + b] : 0.f;
  __syncthreads();
  const float hasmem = (t > 0) ? 1.f : 0.f;
  float mv = 0.f;
  for (int tp = 0; tp < t; ++tp) mv += sh_c[tp] * vals[(size_t)(tp * B_ + b) * VD_ + tid];
  mv *= hasmem;
  // layernorm over V
  red[tid] = mv; __syncthreads();
  for (int off = 64; off > 0; off >>= 1) { if (tid < off) red[tid] += red[tid + off]; __syncthreads(); }
  float mean = red[0] * (1.0f / VD_); __syncthreads();
  float dv = mv - mean;
  red[tid] = dv * dv; __syncthreads();
  for (int off = 64; off > 0; off >>= 1) { if (tid < off) red[tid] += red[tid + off]; __syncthreads(); }
  float var = red[0] * (1.0f / VD_); __syncthreads();
  float m = dv * rsqrtf(var + 1e-5f);
  mem_ws[b * VD_ + tid] = m;
  mem_bf[b * VD_ + tid] = (__bf16)m;
  // write-decision MLP: feat = [max_ip, pos, expl] -> 200 -> 1
  float max_ip = smax * hasmem;
  float pos = (float)t * (1.0f / T_);
  float ex  = expl[t * B_ + b];
  for (int r = tid; r < 200; r += 128) {
    float h = mem_W1[r * 3 + 0] * max_ip + mem_W1[r * 3 + 1] * pos + mem_W1[r * 3 + 2] * ex + mem_b1[r];
    sh_h1[r] = (h > 0.f) ? h : 0.1f * h;
  }
  __syncthreads();
  float p = 0.f;
  for (int r = tid; r < 200; r += 128) p += sh_h1[r] * mem_W2[r];
  red[tid] = p; __syncthreads();
  for (int off = 64; off > 0; off >>= 1) { if (tid < off) red[tid] += red[tid + off]; __syncthreads(); }
  if (tid == 0) probs[t * B_ + b] = 1.0f / (1.0f + expf(-(red[0] + mem_b2[0])));
}

// ---------------- step kernel 4: logits[256,512] = mem @ predWT + pred_b ----------------
__global__ void k_pred_gemm(const __bf16* __restrict__ mem_bf, const __bf16* __restrict__ predWT,
                            const float* __restrict__ pred_b, float* __restrict__ logits_out, int t) {
  const int lane = threadIdx.x & 31, wave = threadIdx.x >> 5;
  const int w = blockIdx.x * 8 + wave;        // 128 waves: 16 m-tiles x 8 n-groups
  const int m0 = (w >> 3) * 16, nt0 = (w & 7) * 4;
  v16bf a[4];
#pragma unroll
  for (int kt = 0; kt < 4; ++kt) a[kt] = load_a_frag(mem_bf, H_, m0, kt * 32, lane);
  const int nn_l = lane & 15;
  const int mb   = m0 + ((lane >> 4) ? 8 : 0);
#pragma unroll
  for (int s = 0; s < 4; ++s) {
    const int nt = nt0 + s;
    v8f acc = {};
#pragma unroll
    for (int kt = 0; kt < 4; ++kt)
      acc = wmma_bf16(a[kt], load_b_swz(predWT, 4, kt, nt, lane), acc);
    const int o = nt * 16 + nn_l;
    const float pb = pred_b[o];
#pragma unroll
    for (int v = 0; v < 8; ++v)
      logits_out[(size_t)(t * B_ + mb + v) * O_ + o] = acc[v] + pb;
  }
}

// ---------------- step kernel 5: preds = softmax(logits) over O ----------------
__global__ void k_softmax(const float* __restrict__ logits, float* __restrict__ preds, int t) {
  __shared__ float red[256];
  const int b = blockIdx.x, tid = threadIdx.x;
  const float* row = logits + (size_t)(t * B_ + b) * O_;
  float x0 = row[tid], x1 = row[tid + 256];
  red[tid] = fmaxf(x0, x1); __syncthreads();
  for (int off = 128; off > 0; off >>= 1) { if (tid < off) red[tid] = fmaxf(red[tid], red[tid + off]); __syncthreads(); }
  float mx = red[0]; __syncthreads();
  float e0 = expf(x0 - mx), e1 = expf(x1 - mx);
  red[tid] = e0 + e1; __syncthreads();
  for (int off = 128; off > 0; off >>= 1) { if (tid < off) red[tid] += red[tid + off]; __syncthreads(); }
  float inv = 1.0f / red[0];
  float* out = preds + (size_t)(t * B_ + b) * O_;
  out[tid] = e0 * inv;
  out[tid + 256] = e1 * inv;
}

// ---------------- step kernel 6: gate GEMM (K=256) fused with hidden update ----------------
__global__ void k_gate(const __bf16* __restrict__ Agate, const __bf16* __restrict__ gateWT,
                       const float* __restrict__ gate_b, const float* __restrict__ pi_ws,
                       const int* __restrict__ obs, const float* __restrict__ embed_W,
                       float* __restrict__ hidden, __bf16* __restrict__ hidden_bf,
                       float* __restrict__ gating_out, float* __restrict__ infh_out, int t) {
  const int lane = threadIdx.x & 31, wave = threadIdx.x >> 5;
  const int w = blockIdx.x * 8 + wave;        // 64 waves: 16 m-tiles x 4 n-groups
  const int m0 = (w >> 2) * 16, nt0 = (w & 3) * 2;
  v16bf a[8];
#pragma unroll
  for (int kt = 0; kt < 8; ++kt) a[kt] = load_a_frag(Agate, 256, m0, kt * 32, lane);
  const int nn_l = lane & 15;
  const int mb   = m0 + ((lane >> 4) ? 8 : 0);
#pragma unroll
  for (int s = 0; s < 2; ++s) {
    const int nt = nt0 + s;
    v8f acc = {};
#pragma unroll
    for (int kt = 0; kt < 8; ++kt)
      acc = wmma_bf16(a[kt], load_b_swz(gateWT, 8, kt, nt, lane), acc);
    const int h = nt * 16 + nn_l;
    const float gb = gate_b[h];
#pragma unroll
    for (int v = 0; v < 8; ++v) {
      int bidx = mb + v;
      float g  = 1.0f / (1.0f + expf(-(acc[v] + gb)));
      float pv = pi_ws[bidx * H_ + h];
      float ih = embed_W[(size_t)obs[t * B_ + bidx] * H_ + h];
      float hn = tanhf(g * pv + (1.0f - g) * ih);
      gating_out[(size_t)(t * B_ + bidx) * H_ + h] = g;
      infh_out[(size_t)(t * B_ + bidx) * H_ + h]   = hn;
      hidden[bidx * H_ + h]    = hn;
      hidden_bf[bidx * H_ + h] = (__bf16)hn;
    }
  }
}

// ---------------- step kernel 7: [keys|vals] = hidden_new @ [WqkT|WvT] ----------------
__global__ void k_kv_gemm(const __bf16* __restrict__ hidden_bf, const __bf16* __restrict__ kvWT,
                          float* __restrict__ keys, float* __restrict__ vals, int t) {
  const int lane = threadIdx.x & 31, wave = threadIdx.x >> 5;
  const int w = blockIdx.x * 8 + wave;        // 48 waves: 16 m-tiles x 3 n-groups
  const int m0 = (w / 3) * 16, nt0 = (w % 3) * 4;
  v16bf a[4];
#pragma unroll
  for (int kt = 0; kt < 4; ++kt) a[kt] = load_a_frag(hidden_bf, H_, m0, kt * 32, lane);
  const int nn_l = lane & 15;
  const int mb   = m0 + ((lane >> 4) ? 8 : 0);
#pragma unroll
  for (int s = 0; s < 4; ++s) {
    const int nt = nt0 + s;
    v8f acc = {};
#pragma unroll
    for (int kt = 0; kt < 4; ++kt)
      acc = wmma_bf16(a[kt], load_b_swz(kvWT, 4, kt, nt, lane), acc);
    const int c = nt * 16 + nn_l;
#pragma unroll
    for (int v = 0; v < 8; ++v) {
      int bidx = mb + v;
      if (c < KD_) keys[(size_t)(t * B_ + bidx) * KD_ + c] = acc[v];
      else         vals[(size_t)(t * B_ + bidx) * VD_ + (c - KD_)] = acc[v];
    }
  }
}

// ---------------- workspace layout (bytes, 256-aligned slots) ----------------
static constexpr size_t alup(size_t x) { return (x + 255) & ~(size_t)255; }
static constexpr size_t OFF_HID   = 0;
static constexpr size_t OFF_HIDB  = alup(OFF_HID   + (size_t)B_ * H_ * 4);
static constexpr size_t OFF_KEYS  = alup(OFF_HIDB  + (size_t)B_ * H_ * 2);
static constexpr size_t OFF_VALS  = alup(OFF_KEYS  + (size_t)T_ * B_ * KD_ * 4);
static constexpr size_t OFF_PROBS = alup(OFF_VALS  + (size_t)T_ * B_ * VD_ * 4);
static constexpr size_t OFF_VEMB  = alup(OFF_PROBS + (size_t)T_ * B_ * 4);
static constexpr size_t OFF_HYP   = alup(OFF_VEMB  + (size_t)T_ * B_ * H_ * 4);
static constexpr size_t OFF_U     = alup(OFF_HYP   + (size_t)T_ * B_ * DM_ * 4);
static constexpr size_t OFF_PI    = alup(OFF_U     + (size_t)B_ * H_ * DM_ * 4);
static constexpr size_t OFF_PIP   = alup(OFF_PI    + (size_t)B_ * H_ * 4);
static constexpr size_t OFF_AGATE = alup(OFF_PIP   + (size_t)B_ * H_ * 4);
static constexpr size_t OFF_MEM   = alup(OFF_AGATE + (size_t)B_ * 256 * 2);
static constexpr size_t OFF_MEMB  = alup(OFF_MEM   + (size_t)B_ * VD_ * 4);
static constexpr size_t OFF_W2P   = alup(OFF_MEMB  + (size_t)B_ * VD_ * 2);
static constexpr size_t OFF_PWT   = alup(OFF_W2P   + (size_t)H_ * H_ * DM_ * 2);
static constexpr size_t OFF_GWT   = alup(OFF_PWT   + (size_t)H_ * O_ * 2);
static constexpr size_t OFF_KVWT  = alup(OFF_GWT   + (size_t)256 * H_ * 2);

extern "C" void kernel_launch(void* const* d_in, const int* in_sizes, int n_in,
                              void* d_out, int out_size, void* d_ws, size_t ws_size,
                              hipStream_t stream) {
  (void)in_sizes; (void)n_in; (void)out_size; (void)ws_size;
  const int*   obs      = (const int*)d_in[0];
  const float* vel      = (const float*)d_in[1];
  const float* expl     = (const float*)d_in[2];
  const float* hid0     = (const float*)d_in[3];
  const float* embed_W  = (const float*)d_in[4];
  const float* vel_W    = (const float*)d_in[5];
  const float* vel_b    = (const float*)d_in[6];
  const float* hyp_W1   = (const float*)d_in[7];
  const float* hyp_b1   = (const float*)d_in[8];
  const float* hyp_W2   = (const float*)d_in[9];
  const float* hyp_b2   = (const float*)d_in[10];
  const float* gate_W   = (const float*)d_in[11];
  const float* gate_b   = (const float*)d_in[12];
  const float* pred_W   = (const float*)d_in[13];
  const float* pred_b   = (const float*)d_in[14];
  const float* mem_W1   = (const float*)d_in[15];
  const float* mem_b1   = (const float*)d_in[16];
  const float* mem_W2   = (const float*)d_in[17];
  const float* mem_b2   = (const float*)d_in[18];
  const float* Wqk      = (const float*)d_in[19];
  const float* Wv       = (const float*)d_in[20];

  char* ws = (char*)d_ws;
  float*  hidden    = (float*)(ws + OFF_HID);
  __bf16* hidden_bf = (__bf16*)(ws + OFF_HIDB);
  float*  keys      = (float*)(ws + OFF_KEYS);
  float*  vals      = (float*)(ws + OFF_VALS);
  float*  probs     = (float*)(ws + OFF_PROBS);
  float*  vemb_all  = (float*)(ws + OFF_VEMB);
  float*  hyp_all   = (float*)(ws + OFF_HYP);
  float*  U         = (float*)(ws + OFF_U);
  float*  pi_ws     = (float*)(ws + OFF_PI);
  float*  pi_pred   = (float*)(ws + OFF_PIP);
  __bf16* Agate     = (__bf16*)(ws + OFF_AGATE);
  float*  mem_ws    = (float*)(ws + OFF_MEM);
  __bf16* mem_bf    = (__bf16*)(ws + OFF_MEMB);
  __bf16* W2p       = (__bf16*)(ws + OFF_W2P);
  __bf16* predWT    = (__bf16*)(ws + OFF_PWT);
  __bf16* gateWT    = (__bf16*)(ws + OFF_GWT);
  __bf16* kvWT      = (__bf16*)(ws + OFF_KVWT);

  const size_t TBO = (size_t)T_ * B_ * O_;
  const size_t TBH = (size_t)T_ * B_ * H_;
  float* out_preds  = (float*)d_out;
  float* out_logits = out_preds + TBO;
  float* out_infh   = out_logits + TBO;
  float* out_gen    = out_infh + TBH;
  float* out_gating = out_gen + TBH;

  // one-time (per call) prep
  k_prep_weights<<<(128 * 4096 + 128 * 512 + 256 * 128 + 128 * 192 + 255) / 256, 256, 0, stream>>>(
      hyp_W2, pred_W, gate_W, Wqk, Wv, W2p, predWT, gateWT, kvWT);
  k_prep_state<<<T_ * B_, H_, 0, stream>>>(vel, vel_W, vel_b, hyp_W1, hyp_b1, hid0,
                                           vemb_all, hyp_all, hidden, hidden_bf);

  for (int t = 0; t < T_; ++t) {
    k_hyper_gemm<<<128, 256, 0, stream>>>(hidden_bf, W2p, U);
    k_pi<<<B_, H_, 0, stream>>>(U, hyp_all, hidden, hyp_b2, vemb_all, obs, embed_W,
                                pi_ws, pi_pred, Agate, out_gen, t);
    k_attn<<<B_, 128, 0, stream>>>(pi_pred, Wqk, keys, vals, probs, expl,
                                   mem_W1, mem_b1, mem_W2, mem_b2, mem_ws, mem_bf, t);
    k_pred_gemm<<<16, 256, 0, stream>>>(mem_bf, predWT, pred_b, out_logits, t);
    k_softmax<<<B_, 256, 0, stream>>>(out_logits, out_preds, t);
    k_gate<<<8, 256, 0, stream>>>(Agate, gateWT, gate_b, pi_ws, obs, embed_W,
                                  hidden, hidden_bf, out_gating, out_infh, t);
    k_kv_gemm<<<6, 256, 0, stream>>>(hidden_bf, kvWT, keys, vals, t);
  }
}
